// SelfAttention_59261958750565
// MI455X (gfx1250) — compile-verified, matrix operand order checked by reading, not run on hardware
//
#include <hip/hip_runtime.h>

typedef __attribute__((ext_vector_type(16))) __bf16 v16bf;
typedef __attribute__((ext_vector_type(8)))  float  v8f;
typedef __attribute__((ext_vector_type(4)))  int    v4i;

union FragU { v16bf bf; uint4 u[2]; };

__device__ __forceinline__ unsigned short f2bf(float x) {
  unsigned int u = __float_as_uint(x);
  u += 0x7FFFu + ((u >> 16) & 1u);          // round-to-nearest-even
  return (unsigned short)(u >> 16);
}
__device__ __forceinline__ float bf2f(unsigned short h) {
  return __uint_as_float(((unsigned int)h) << 16);
}

// ---- CDNA5 async global->LDS (ASYNCcnt) with safe fallback -------------
#if __has_builtin(__builtin_amdgcn_global_load_async_to_lds_b128)
#define ATHENA_HAVE_ASYNC_LDS 1
#else
#define ATHENA_HAVE_ASYNC_LDS 0
#endif

typedef __attribute__((address_space(1))) v4i* as1_v4i_p;
typedef __attribute__((address_space(3))) v4i* as3_v4i_p;

__device__ __forceinline__ void async_cp16(const unsigned short* g, unsigned short* l) {
#if ATHENA_HAVE_ASYNC_LDS
  // generic->AS1 is bit-identical; generic LDS->AS3 is the low 32 bits.
  __builtin_amdgcn_global_load_async_to_lds_b128(
      (as1_v4i_p)(unsigned long long)g,
      (as3_v4i_p)(unsigned int)(unsigned long long)l, 0, 0);
#else
  *(uint4*)l = *(const uint4*)g;
#endif
}
__device__ __forceinline__ void async_wait() {
#if ATHENA_HAVE_ASYNC_LDS
  asm volatile("s_wait_asynccnt 0" ::: "memory");
#endif
}

// Load one 16x32 (A) or NxK (B) bf16 WMMA fragment from a row-major
// [rows][ld] uint16 tile. Layout per CDNA5 ISA 7.12.2:
//   lanes 0-15 : row = ln, K = koff+0..7  (VGPR0-3), koff+16..23 (VGPR4-7)
//   lanes 16-31: row = ln, K = koff+8..15 (VGPR0-3), koff+24..31 (VGPR4-7)
__device__ __forceinline__ v16bf load_frag(const unsigned short* p, int row0, int ld, int koff) {
  const int lane = threadIdx.x & 31;
  const int half = lane >> 4, ln = lane & 15;
  const unsigned short* q = p + (size_t)(row0 + ln) * ld + koff + half * 8;
  FragU f;
  f.u[0] = *(const uint4*)(q);
  f.u[1] = *(const uint4*)(q + 16);
  return f.bf;
}

__device__ __forceinline__ float rmax16(float v) {
  #pragma unroll
  for (int m = 1; m < 16; m <<= 1) v = fmaxf(v, __shfl_xor(v, m, 32));
  return v;   // masks 1,2,4,8 never cross the 16-lane half
}
__device__ __forceinline__ float rsum16(float v) {
  #pragma unroll
  for (int m = 1; m < 16; m <<= 1) v += __shfl_xor(v, m, 32);
  return v;
}

// ---------------------------------------------------------------- convert
__global__ void f32_to_bf16_kernel(const float* __restrict__ x,
                                   unsigned short* __restrict__ y, size_t n) {
  size_t i = (size_t)blockIdx.x * blockDim.x + threadIdx.x;
  size_t st = (size_t)gridDim.x * blockDim.x;
  for (; i < n; i += st) y[i] = f2bf(x[i]);
}

// ---------------------------------------------------------------- GEMM
// C[M,N] = A[M,K] (bf16, row-major) @ B[N,K]^T (bf16, row-major) + bias[N]
// Block: 256 thr = 8 waves (2x4), tile 128x256, wave tile 64x64 (16 WMMA/step).
// K staged in 32-chunks, double-buffered LDS fed by async global->LDS.
template <bool OUT_F32>
__global__ __launch_bounds__(256)
void gemm_bf16_kernel(const unsigned short* __restrict__ A,
                      const unsigned short* __restrict__ Bm,
                      const float* __restrict__ bias,
                      void* __restrict__ Cout, int M, int N, int K) {
  __shared__ unsigned short sA[2][128 * 32];   // 16 KB
  __shared__ unsigned short sB[2][256 * 32];   // 32 KB
  const int tid = threadIdx.x;
  const int wid = tid >> 5;
  const int wm = wid >> 2, wn = wid & 3;       // 2x4 wave grid
  const int m0 = blockIdx.y * 128, n0 = blockIdx.x * 256;
  const int KS = K >> 5;

  auto stage = [&](int ks, int buf) {
    const size_t k0 = (size_t)ks * 32;
    #pragma unroll
    for (int i = 0; i < 2; ++i) {              // A: 512 b128 segments
      const int seg = tid + (i << 8);
      const int row = seg >> 2, kk = (seg & 3) << 3;
      async_cp16(&A[(size_t)(m0 + row) * K + k0 + kk], &sA[buf][row * 32 + kk]);
    }
    #pragma unroll
    for (int i = 0; i < 4; ++i) {              // B: 1024 b128 segments
      const int seg = tid + (i << 8);
      const int row = seg >> 2, kk = (seg & 3) << 3;
      async_cp16(&Bm[(size_t)(n0 + row) * K + k0 + kk], &sB[buf][row * 32 + kk]);
    }
  };

  v8f acc[4][4];
  const v8f vz = {0.f,0.f,0.f,0.f,0.f,0.f,0.f,0.f};
  #pragma unroll
  for (int mt = 0; mt < 4; ++mt)
    #pragma unroll
    for (int nt = 0; nt < 4; ++nt) acc[mt][nt] = vz;

  stage(0, 0);
  async_wait();
  __syncthreads();

  for (int ks = 0; ks < KS; ++ks) {
    const int buf = ks & 1;
    if (ks + 1 < KS) {
      stage(ks + 1, buf ^ 1);                  // fills other buffer, no wait yet
      if (ks + 2 < KS) {                       // L2 prefetch (global_prefetch_b8)
        __builtin_prefetch(&A [(size_t)(m0 + (tid >> 1)) * K + (size_t)(ks + 2) * 32], 0, 1);
        __builtin_prefetch(&Bm[(size_t)(n0 + tid)        * K + (size_t)(ks + 2) * 32], 0, 1);
      }
    }
    v16bf bfr[4];
    #pragma unroll
    for (int nt = 0; nt < 4; ++nt)
      bfr[nt] = load_frag(sB[buf], wn * 64 + nt * 16, 32, 0);
    #pragma unroll
    for (int mt = 0; mt < 4; ++mt) {
      v16bf af = load_frag(sA[buf], wm * 64 + mt * 16, 32, 0);
      #pragma unroll
      for (int nt = 0; nt < 4; ++nt)
        acc[mt][nt] = __builtin_amdgcn_wmma_f32_16x16x32_bf16(
            false, af, false, bfr[nt], (short)0, acc[mt][nt], false, false);
    }
    if (ks + 1 == KS) break;
    async_wait();
    __syncthreads();
  }

  const int lane = tid & 31, half = lane >> 4, ln = lane & 15;
  #pragma unroll
  for (int mt = 0; mt < 4; ++mt)
    #pragma unroll
    for (int nt = 0; nt < 4; ++nt) {
      const int col = n0 + wn * 64 + nt * 16 + ln;
      const float bv = bias[col];
      #pragma unroll
      for (int r = 0; r < 8; ++r) {
        const int row = m0 + wm * 64 + mt * 16 + r + 8 * half;
        const float v = acc[mt][nt][r] + bv;
        if (OUT_F32) ((float*)Cout)[(size_t)row * N + col] = v;
        else ((unsigned short*)Cout)[(size_t)row * N + col] = f2bf(v);
      }
    }
}

// ---------------------------------------------------------------- RoPE + split
// qkv bf16 (B,S,3E) -> q_pad/k_pad bf16 (B,H,S,96) (D padded 80->96, rotary
// applied), vT bf16 (B,H,D,S) (transposed so PV B-fragments are K-contiguous).
__global__ __launch_bounds__(128)
void rope_split_kernel(const unsigned short* __restrict__ qkv,
                       const float* __restrict__ cosd, const float* __restrict__ sind,
                       unsigned short* __restrict__ qp, unsigned short* __restrict__ kp,
                       unsigned short* __restrict__ vt) {
  const int s = blockIdx.x, h = blockIdx.y, b = blockIdx.z;
  const int d = threadIdx.x;
  if (d >= 96) return;
  const size_t bh = (size_t)b * 32 + h;
  const size_t qrow = (bh * 2048 + s) * 96;
  if (d >= 80) { qp[qrow + d] = 0; kp[qrow + d] = 0; return; }
  const size_t base = ((size_t)b * 2048 + s) * 7680 + h * 80 + d;
  const float cv = cosd[s * 80 + d], sv = sind[s * 80 + d];
  const float q0 = bf2f(qkv[base]);
  const float k0 = bf2f(qkv[base + 2560]);
  const float v0 = bf2f(qkv[base + 5120]);
  float qr, kr;
  if (d < 40) {
    qr = q0 * cv - bf2f(qkv[base + 40])        * sv;
    kr = k0 * cv - bf2f(qkv[base + 2560 + 40]) * sv;
  } else {
    qr = q0 * cv + bf2f(qkv[base - 40])        * sv;
    kr = k0 * cv + bf2f(qkv[base + 2560 - 40]) * sv;
  }
  qp[qrow + d] = f2bf(qr);
  kp[qrow + d] = f2bf(kr);
  vt[(bh * 80 + d) * 2048 + s] = f2bf(v0);
}

// ---------------------------------------------------------------- attention
// Flash-style: block = 64 queries (4 waves x 16 rows), online softmax over
// 64-key chunks. Scores: 12 wmma/chunk/wave; PV: 10 wmma/chunk/wave.
__global__ __launch_bounds__(128)
void attn_kernel(const unsigned short* __restrict__ qp,
                 const unsigned short* __restrict__ kp,
                 const unsigned short* __restrict__ vt,
                 unsigned short* __restrict__ ctx) {
  constexpr int S = 2048, H = 32, E = 2560;
  const int qb = blockIdx.x, h = blockIdx.y, b = blockIdx.z;
  const int tid = threadIdx.x;
  const int w = tid >> 5, lane = tid & 31, half = lane >> 4, ln = lane & 15;
  const size_t bh = (size_t)b * H + h;

  __shared__ unsigned short sK[64 * 96];     // 64 keys x Dpad
  __shared__ unsigned short sV[80 * 64];     // D x 64 keys (transposed)
  __shared__ unsigned short sP[4][16 * 64];  // per-wave probs staging

  // q fragments for this wave's 16 rows, K = 96 (3 chunks of 32)
  const unsigned short* qbase = qp + (bh * S + (size_t)qb * 64 + w * 16) * 96;
  v16bf qf[3];
  #pragma unroll
  for (int kc = 0; kc < 3; ++kc) qf[kc] = load_frag(qbase, 0, 96, kc * 32);

  float m_run[8], l_run[8];
  v8f oacc[5];
  const v8f vz = {0.f,0.f,0.f,0.f,0.f,0.f,0.f,0.f};
  #pragma unroll
  for (int r = 0; r < 8; ++r) { m_run[r] = -1e30f; l_run[r] = 0.f; }
  #pragma unroll
  for (int t = 0; t < 5; ++t) oacc[t] = vz;

  const float scale = 0.11180339887498949f;  // 1/sqrt(80)

  for (int kb = 0; kb < S / 64; ++kb) {
    // stage K tile (64x96 halves = 768 b128 segs) + V^T tile (80x64 = 640)
    const unsigned short* kgb = kp + (bh * S + (size_t)kb * 64) * 96;
    #pragma unroll
    for (int i = 0; i < 6; ++i) {
      const int seg = tid + i * 128;
      const int row = seg / 12, kk = (seg % 12) * 8;
      async_cp16(&kgb[row * 96 + kk], &sK[row * 96 + kk]);
    }
    const unsigned short* vgb = vt + bh * 80 * S + (size_t)kb * 64;
    #pragma unroll
    for (int i = 0; i < 5; ++i) {
      const int seg = tid + i * 128;
      const int row = seg >> 3, kk = (seg & 7) * 8;
      async_cp16(&vgb[(size_t)row * S + kk], &sV[row * 64 + kk]);
    }
    async_wait();
    __syncthreads();

    // scores: 4 key tiles x (K=96 -> 3 wmma)
    v8f sc[4];
    #pragma unroll
    for (int nt = 0; nt < 4; ++nt) {
      sc[nt] = vz;
      #pragma unroll
      for (int kc = 0; kc < 3; ++kc)
        sc[nt] = __builtin_amdgcn_wmma_f32_16x16x32_bf16(
            false, qf[kc], false, load_frag(sK, nt * 16, 96, kc * 32),
            (short)0, sc[nt], false, false);
    }

    // online softmax: row r+8*half lives in VGPR r across the 16-lane half
    #pragma unroll
    for (int r = 0; r < 8; ++r) {
      float mx = sc[0][r];
      #pragma unroll
      for (int nt = 1; nt < 4; ++nt) mx = fmaxf(mx, sc[nt][r]);
      mx = rmax16(mx) * scale;
      const float mnew = fmaxf(m_run[r], mx);
      const float corr = __expf(m_run[r] - mnew);
      float sum = 0.f;
      #pragma unroll
      for (int nt = 0; nt < 4; ++nt) {
        const float p = __expf(sc[nt][r] * scale - mnew);
        sc[nt][r] = p;
        sum += p;
      }
      sum = rsum16(sum);
      l_run[r] = l_run[r] * corr + sum;
      m_run[r] = mnew;
      #pragma unroll
      for (int t = 0; t < 5; ++t) oacc[t][r] *= corr;
    }

    // stage probs (C layout -> row-major LDS) for A-fragment reload
    unsigned short* pw = &sP[w][0];
    #pragma unroll
    for (int nt = 0; nt < 4; ++nt)
      #pragma unroll
      for (int r = 0; r < 8; ++r)
        pw[(r + 8 * half) * 64 + nt * 16 + ln] = f2bf(sc[nt][r]);
    asm volatile("s_wait_dscnt 0" ::: "memory");  // wave-local LDS RAW

    // PV: 2 key sub-chunks x 5 dim tiles
    #pragma unroll
    for (int hc = 0; hc < 2; ++hc) {
      v16bf pf = load_frag(pw, 0, 64, hc * 32);
      #pragma unroll
      for (int t = 0; t < 5; ++t)
        oacc[t] = __builtin_amdgcn_wmma_f32_16x16x32_bf16(
            false, pf, false, load_frag(sV, t * 16, 64, hc * 32),
            (short)0, oacc[t], false, false);
    }
    __syncthreads();
  }

  // epilogue: normalize, write ctx as (b,s,h*80+d) bf16
  #pragma unroll
  for (int r = 0; r < 8; ++r) {
    const int s = qb * 64 + w * 16 + r + 8 * half;
    const float inv = 1.0f / l_run[r];
    #pragma unroll
    for (int t = 0; t < 5; ++t)
      ctx[((size_t)b * S + s) * E + h * 80 + t * 16 + ln] = f2bf(oacc[t][r] * inv);
  }
}

// ---------------------------------------------------------------- launch
extern "C" void kernel_launch(void* const* d_in, const int* in_sizes, int n_in,
                              void* d_out, int out_size, void* d_ws, size_t ws_size,
                              hipStream_t stream) {
  const float* hidden = (const float*)d_in[0];
  const float* wqkv   = (const float*)d_in[1];
  const float* bqkv   = (const float*)d_in[2];
  const float* wout   = (const float*)d_in[3];
  const float* bout   = (const float*)d_in[4];
  const float* cosd   = (const float*)d_in[5];
  const float* sind   = (const float*)d_in[6];
  float* out = (float*)d_out;

  const size_t M = 4096, E = 2560, N3 = 7680;
  char* ws = (char*)d_ws;
  unsigned short* hbf  = (unsigned short*)ws; ws += M * E * 2;
  unsigned short* wqbf = (unsigned short*)ws; ws += N3 * E * 2;
  unsigned short* wobf = (unsigned short*)ws; ws += E * E * 2;
  unsigned short* qkv  = (unsigned short*)ws; ws += M * N3 * 2;
  unsigned short* qpad = (unsigned short*)ws; ws += (size_t)2 * 32 * 2048 * 96 * 2;
  unsigned short* kpad = (unsigned short*)ws; ws += (size_t)2 * 32 * 2048 * 96 * 2;
  unsigned short* vT   = (unsigned short*)ws; ws += (size_t)2 * 32 * 2048 * 80 * 2;
  unsigned short* ctx  = (unsigned short*)ws;

  f32_to_bf16_kernel<<<4096, 256, 0, stream>>>(hidden, hbf, M * E);
  f32_to_bf16_kernel<<<4096, 256, 0, stream>>>(wqkv, wqbf, N3 * E);
  f32_to_bf16_kernel<<<4096, 256, 0, stream>>>(wout, wobf, E * E);

  gemm_bf16_kernel<false><<<dim3(N3 / 256, M / 128), 256, 0, stream>>>(
      hbf, wqbf, bqkv, qkv, (int)M, (int)N3, (int)E);

  rope_split_kernel<<<dim3(2048, 32, 2), 128, 0, stream>>>(qkv, cosd, sind, qpad, kpad, vT);

  attn_kernel<<<dim3(32, 32, 2), 128, 0, stream>>>(qpad, kpad, vT, ctx);

  gemm_bf16_kernel<true><<<dim3(E / 256, M / 128), 256, 0, stream>>>(
      ctx, wobf, bout, out, (int)M, (int)E, (int)E);
}
